// MHSelfAttn_5875515261378
// MI455X (gfx1250) — matrix operand order repaired and checked by reading, not compiled, not run on hardware
//
#include <hip/hip_runtime.h>
#include <math.h>

// Problem constants (match reference)
#define B_   4
#define S_   1024
#define DM_  512
#define H_   8
#define DK_  64
#define KP1_ 16
#define W2_  32
#define M_   (B_ * S_)            // 4096 GEMM rows
#define HS_  (B_ * H_ * S_ * DK_) // 2,097,152 floats per head tensor

#define MT_    (M_ / 16)          // 256 M-tiles of 16
#define KT_    (DM_ / 32)         // 16 K-tiles of 32
#define NT_    (DM_ / 16)         // 32 N-tiles of 16
#define AFRAGS (MT_ * KT_ * 32)   // fragments per packed activation (131072)
#define BFRAGS (NT_ * KT_ * 32)   // fragments per packed weight (16384)

typedef __attribute__((ext_vector_type(16))) __bf16 v16bf;
typedef __attribute__((ext_vector_type(8)))  float  v8f;

// ---------------------------------------------------------------------------
// Pack kernels: convert fp32 -> bf16 and pre-swizzle into WMMA fragment-major
// layout (CDNA5 ISA 7.12.2 wave32), so the GEMM inner loop is load+wmma only.
// A 16x32 (MxK): lane L<16 -> row=L, VGPR j holds K = 16*(j>>2)+2*(j&3) (+1);
//                lanes>=16 shifted +8 in K.
// Fragment index: ((mtile*KT + ktile)*32 + lane), 16 bf16 (32 B) each.
// ---------------------------------------------------------------------------
__global__ __launch_bounds__(256) void pack_a_kernel(const float* __restrict__ X,
                                                     __bf16* __restrict__ out) {
    const int t    = blockIdx.x * 256 + threadIdx.x;  // 131072 threads
    const int lane = t & 31;
    const int kt   = (t >> 5) & (KT_ - 1);
    const int mt   = t >> 9;
    const int m    = (mt << 4) + (lane & 15);
    const int kb   = (kt << 5) + ((lane >> 4) << 3);
    const float* src = X + (size_t)m * DM_ + kb;
    v16bf f;
#pragma unroll
    for (int j = 0; j < 8; ++j) {
        int k = ((j >> 2) << 4) | ((j & 3) << 1);
        float2 v = *reinterpret_cast<const float2*>(src + k);
        f[2 * j]     = (__bf16)v.x;
        f[2 * j + 1] = (__bf16)v.y;
    }
    *reinterpret_cast<v16bf*>(out + (size_t)t * 16) = f;  // coalesced 32B/lane
}

// Heads tensor [b,h,s,dk] viewed as A[m=b*S+s][k=h*64+dk]; a 32-wide K tile
// stays inside one head, so h is constant per fragment.
__global__ __launch_bounds__(256) void pack_heads_kernel(const float* __restrict__ heads,
                                                         __bf16* __restrict__ out) {
    const int t    = blockIdx.x * 256 + threadIdx.x;
    const int lane = t & 31;
    const int kt   = (t >> 5) & (KT_ - 1);
    const int mt   = t >> 9;
    const int m    = (mt << 4) + (lane & 15);
    const int bi   = m >> 10, s = m & (S_ - 1);
    const int h    = kt >> 1;
    const int dkb  = ((kt & 1) << 5) + ((lane >> 4) << 3);
    const float* src = heads + ((size_t)(bi * H_ + h) * S_ + s) * DK_ + dkb;
    v16bf f;
#pragma unroll
    for (int j = 0; j < 8; ++j) {
        int k = ((j >> 2) << 4) | ((j & 3) << 1);
        float2 v = *reinterpret_cast<const float2*>(src + k);
        f[2 * j]     = (__bf16)v.x;
        f[2 * j + 1] = (__bf16)v.y;
    }
    *reinterpret_cast<v16bf*>(out + (size_t)t * 16) = f;
}

// B 32x16 (KxN) from row-major W[N][K]: lane -> col, lanes<16 hold K=0..7,16..23; lanes>=16 hold K=8..15,24..31.
// Packs all four weights (Wq,Wk,Wv,Wo) in one launch.
__global__ __launch_bounds__(256) void pack_b_kernel(
    const float* __restrict__ W0, const float* __restrict__ W1,
    const float* __restrict__ W2, const float* __restrict__ W3,
    __bf16* __restrict__ out) {
    const int t     = blockIdx.x * 256 + threadIdx.x;  // 4 * 16384 threads
    const int lane  = t & 31;
    const int kt    = (t >> 5) & (KT_ - 1);
    const int nt    = (t >> 9) & (NT_ - 1);
    const int which = t >> 14;
    const float* W = (which == 0) ? W0 : (which == 1) ? W1 : (which == 2) ? W2 : W3;
    const int n  = (nt << 4) + (lane & 15);
    const int kb = (kt << 5) + ((lane >> 4) << 3);
    const float4* p = reinterpret_cast<const float4*>(W + (size_t)n * DM_ + kb);
    v16bf f;
#pragma unroll
    for (int i = 0; i < 4; ++i) {
        float4 v = p[i + (i & 2)];
        f[4 * i + 0] = (__bf16)v.x;
        f[4 * i + 1] = (__bf16)v.y;
        f[4 * i + 2] = (__bf16)v.z;
        f[4 * i + 3] = (__bf16)v.w;
    }
    *reinterpret_cast<v16bf*>(out + (size_t)t * 16) = f;
}

// ---------------------------------------------------------------------------
// Fragment fetch helpers (32 B contiguous vector loads).
// ---------------------------------------------------------------------------
__device__ __forceinline__ v16bf afrag(const __bf16* __restrict__ Ap, int mt, int kt, int lane) {
    return *reinterpret_cast<const v16bf*>(Ap + (((size_t)mt * KT_ + kt) * 32 + lane) * 16);
}
__device__ __forceinline__ v16bf bfrag(const __bf16* __restrict__ Bp, int nt, int kt, int lane) {
    return *reinterpret_cast<const v16bf*>(Bp + (((size_t)nt * KT_ + kt) * 32 + lane) * 16);
}

// ---------------------------------------------------------------------------
// Kernel: fused q/k/v projections from packed operands.
// One wave -> 32x64 output block (2 M-tiles x 4 N-tiles, 8 accumulators).
// ---------------------------------------------------------------------------
__global__ __launch_bounds__(256) void proj_gemm_kernel(
    const __bf16* __restrict__ ApQ, const __bf16* __restrict__ ApK,
    const __bf16* __restrict__ ApV, const __bf16* __restrict__ Bpack,
    float* __restrict__ q_ws, float* __restrict__ k_ws, float* __restrict__ v_ws) {
    const int lane = threadIdx.x & 31;
    const int wave = threadIdx.x >> 5;
    const int job  = blockIdx.x * 8 + wave;   // 3 * 1024 = 3072 jobs
    const int which = job >> 10;
    const int rem   = job & 1023;
    const int mb = rem >> 3;                   // 128 blocks of 32 rows
    const int nb = rem & 7;                    // 8 strips of 64 cols

    const __bf16* Ap = (which == 0) ? ApQ : (which == 1) ? ApK : ApV;
    const __bf16* Bp = Bpack + (size_t)which * (BFRAGS * 16);
    float*        O  = (which == 0) ? q_ws : (which == 1) ? k_ws : v_ws;

    v8f acc[2][4] = {};
    for (int kt = 0; kt < KT_; ++kt) {
        v16bf a0 = afrag(Ap, 2 * mb + 0, kt, lane);
        v16bf a1 = afrag(Ap, 2 * mb + 1, kt, lane);
#pragma unroll
        for (int t = 0; t < 4; ++t) {
            v16bf b = bfrag(Bp, nb * 4 + t, kt, lane);
            acc[0][t] = __builtin_amdgcn_wmma_f32_16x16x32_bf16(
                false, a0, false, b, (short)0, acc[0][t], false, false);
            acc[1][t] = __builtin_amdgcn_wmma_f32_16x16x32_bf16(
                false, a1, false, b, (short)0, acc[1][t], false, false);
        }
    }
    // C layout: VGPR r -> row r (+8 for lanes>=16), col = lane&15.
    const int rofs = (lane >> 4) << 3;
#pragma unroll
    for (int u = 0; u < 2; ++u) {
#pragma unroll
        for (int r = 0; r < 8; ++r) {
            const int m  = (2 * mb + u) * 16 + r + rofs;
            const int bi = m >> 10, s = m & (S_ - 1);
#pragma unroll
            for (int t = 0; t < 4; ++t) {
                const int n   = nb * 64 + t * 16 + (lane & 15);
                const int h   = n >> 6, dki = n & (DK_ - 1);
                O[((size_t)(bi * H_ + h) * S_ + s) * DK_ + dki] = acc[u][t][r];
            }
        }
    }
}

// ---------------------------------------------------------------------------
// Kernel: gathered scores. score = (q[qi].k[ki] + q[qi].rel_k + rel_q.k[ki])/24
// ---------------------------------------------------------------------------
__global__ __launch_bounds__(256) void scores_kernel(
    const int* __restrict__ pos_enc, const int* __restrict__ pos_pad,
    const float* __restrict__ rel_q, const float* __restrict__ rel_k,
    const float* __restrict__ q_ws,  const float* __restrict__ k_ws,
    float* __restrict__ scores) {
    const int idx = blockIdx.x * 256 + threadIdx.x;  // [bh, w, s]
    const int s  = idx & (S_ - 1);
    const int w  = (idx >> 10) & (W2_ - 1);
    const int bh = idx >> 15;
    const int h  = bh & (H_ - 1);

    const int pidx = (bh * KP1_ + (w & 15)) * S_ + s;
    const int penc = pos_enc[pidx];
    const int ppad = pos_pad[pidx];
    const bool neg = (penc == -1);
    const int  pe  = neg ? 0 : penc;
    const int  qi  = (w < KP1_) ? ppad : pe;
    const int  ki  = (w < KP1_) ? pe : ppad;
    const bool masked = neg || (w == 0);  // mask_flat[:, :, :s] forced True

    float sc = -1e9f;
    if (!masked) {
        const float4* qr = (const float4*)(q_ws + (size_t)(bh * S_ + qi) * DK_);
        const float4* kr = (const float4*)(k_ws + (size_t)(bh * S_ + ki) * DK_);
        const float4* rk = (const float4*)(rel_k + (size_t)(h * W2_ + w) * DK_);
        const float4* rq = (const float4*)(rel_q + (size_t)(h * W2_ + w) * DK_);
        float acc = 0.f;
#pragma unroll
        for (int d = 0; d < DK_ / 4; ++d) {
            float4 qv = qr[d], kv = kr[d], kva = rk[d], qva = rq[d];
            acc += qv.x * kv.x + qv.y * kv.y + qv.z * kv.z + qv.w * kv.w;     // c2c
            acc += qv.x * kva.x + qv.y * kva.y + qv.z * kva.z + qv.w * kva.w; // c2p
            acc += qva.x * kv.x + qva.y * kv.y + qva.z * kv.z + qva.w * kv.w; // p2c
        }
        sc = acc * (1.0f / 24.0f);  // 3*sqrt(64)
    }
    scores[idx] = sc;
}

// ---------------------------------------------------------------------------
// Kernel: softmax over flattened w2*s = 32768 axis; one block per (b,h).
// ---------------------------------------------------------------------------
__global__ __launch_bounds__(256) void softmax_kernel(float* __restrict__ scores) {
    __shared__ float red[256];
    float* p = scores + (size_t)blockIdx.x * (W2_ * S_);
    const int N = W2_ * S_, tid = threadIdx.x;

    float m = -1e30f;
    for (int i = tid; i < N; i += 256) m = fmaxf(m, p[i]);
    red[tid] = m; __syncthreads();
    for (int o = 128; o > 0; o >>= 1) {
        if (tid < o) red[tid] = fmaxf(red[tid], red[tid + o]);
        __syncthreads();
    }
    m = red[0]; __syncthreads();

    float sum = 0.f;
    for (int i = tid; i < N; i += 256) sum += expf(p[i] - m);
    red[tid] = sum; __syncthreads();
    for (int o = 128; o > 0; o >>= 1) {
        if (tid < o) red[tid] += red[tid + o];
        __syncthreads();
    }
    const float inv = 1.0f / red[0];

    for (int i = tid; i < N; i += 256) p[i] = expf(p[i] - m) * inv;
}

// ---------------------------------------------------------------------------
// Kernel: scatter-add heads[b,h,qi,:] += prob * (v[b,h,ki,:] + rel_v[h,w,:])
// ---------------------------------------------------------------------------
__global__ __launch_bounds__(256) void zero_kernel(float* __restrict__ p, int n) {
    int i = blockIdx.x * 256 + threadIdx.x;
    if (i < n) p[i] = 0.0f;
}

__global__ __launch_bounds__(256) void scatter_kernel(
    const float* __restrict__ probs, const int* __restrict__ pos_enc,
    const int* __restrict__ pos_pad, const float* __restrict__ v_ws,
    const float* __restrict__ rel_v, float* __restrict__ heads) {
    const unsigned gid = blockIdx.x * 256u + threadIdx.x; // slot*64 + dk
    const int dki = (int)(gid & (DK_ - 1));
    const int idx = (int)(gid >> 6);
    const float prob = probs[idx];
    if (prob == 0.0f) return;   // masked slots underflow to exactly 0

    const int s  = idx & (S_ - 1);
    const int w  = (idx >> 10) & (W2_ - 1);
    const int bh = idx >> 15;
    const int h  = bh & (H_ - 1);
    const int pidx = (bh * KP1_ + (w & 15)) * S_ + s;
    const int penc = pos_enc[pidx];
    const int ppad = pos_pad[pidx];
    const int pe   = (penc == -1) ? 0 : penc;
    const int qi   = (w < KP1_) ? ppad : pe;
    const int ki   = (w < KP1_) ? pe : ppad;

    const float vv = v_ws[(size_t)(bh * S_ + ki) * DK_ + dki] +
                     rel_v[(size_t)(h * W2_ + w) * DK_ + dki];
    __hip_atomic_fetch_add(heads + (size_t)(bh * S_ + qi) * DK_ + dki,
                           prob * vv, __ATOMIC_RELAXED, __HIP_MEMORY_SCOPE_AGENT);
}

// ---------------------------------------------------------------------------
// Kernel: output projection from packed heads. out = heads[b,s,h*dk] @ Wo^T.
// ---------------------------------------------------------------------------
__global__ __launch_bounds__(256) void out_gemm_kernel(
    const __bf16* __restrict__ ApH, const __bf16* __restrict__ BpO,
    float* __restrict__ out) {
    const int lane = threadIdx.x & 31;
    const int wave = threadIdx.x >> 5;
    const int job  = blockIdx.x * 8 + wave;   // 1024 jobs
    const int mb = job >> 3;
    const int nb = job & 7;

    v8f acc[2][4] = {};
    for (int kt = 0; kt < KT_; ++kt) {
        v16bf a0 = afrag(ApH, 2 * mb + 0, kt, lane);
        v16bf a1 = afrag(ApH, 2 * mb + 1, kt, lane);
#pragma unroll
        for (int t = 0; t < 4; ++t) {
            v16bf b = bfrag(BpO, nb * 4 + t, kt, lane);
            acc[0][t] = __builtin_amdgcn_wmma_f32_16x16x32_bf16(
                false, a0, false, b, (short)0, acc[0][t], false, false);
            acc[1][t] = __builtin_amdgcn_wmma_f32_16x16x32_bf16(
                false, a1, false, b, (short)0, acc[1][t], false, false);
        }
    }
    const int rofs = (lane >> 4) << 3;
#pragma unroll
    for (int u = 0; u < 2; ++u) {
#pragma unroll
        for (int r = 0; r < 8; ++r) {
            const int m = (2 * mb + u) * 16 + r + rofs;
#pragma unroll
            for (int t = 0; t < 4; ++t) {
                const int n = nb * 64 + t * 16 + (lane & 15);
                out[(size_t)m * DM_ + n] = acc[u][t][r];
            }
        }
    }
}

// ---------------------------------------------------------------------------
extern "C" void kernel_launch(void* const* d_in, const int* in_sizes, int n_in,
                              void* d_out, int out_size, void* d_ws, size_t ws_size,
                              hipStream_t stream) {
    const float* query   = (const float*)d_in[0];
    const float* key     = (const float*)d_in[1];
    const float* value   = (const float*)d_in[2];
    const int*   pos_enc = (const int*)  d_in[3];
    const int*   pos_pad = (const int*)  d_in[4];
    const float* rel_q   = (const float*)d_in[5];
    const float* rel_k   = (const float*)d_in[6];
    const float* rel_v   = (const float*)d_in[7];
    const float* Wq      = (const float*)d_in[8];
    const float* Wk      = (const float*)d_in[9];
    const float* Wv      = (const float*)d_in[10];
    const float* Wo      = (const float*)d_in[11];
    float*       out     = (float*)d_out;

    // Workspace layout (50 MB total)
    float* ws     = (float*)d_ws;
    float* q_ws   = ws;                    // 8 MB fp32 [b,h,s,dk]
    float* k_ws   = ws + (size_t)HS_;      // 8 MB
    float* v_ws   = ws + 2 * (size_t)HS_;  // 8 MB
    float* heads  = ws + 3 * (size_t)HS_;  // 8 MB (scatter accumulator)
    float* scores = ws + 4 * (size_t)HS_;  // 4 MB (B*H*W2*S floats)
    __bf16* packs = (__bf16*)(ws + 4 * (size_t)HS_ + B_ * H_ * W2_ * S_);
    __bf16* ApQ   = packs;                          // 4 MB each (AFRAGS*16 bf16)
    __bf16* ApK   = packs + (size_t)AFRAGS * 16;
    __bf16* ApV   = packs + 2 * (size_t)AFRAGS * 16;
    __bf16* Bpack = packs + 3 * (size_t)AFRAGS * 16; // 4 weights * 512 KB
    __bf16* ApH   = ApQ;  // reuse: Q pack dead after proj GEMM

    const int nslots = B_ * H_ * W2_ * S_;  // 1,048,576

    zero_kernel<<<(HS_ + 255) / 256, 256, 0, stream>>>(heads, HS_);
    pack_a_kernel<<<AFRAGS / 256, 256, 0, stream>>>(query, ApQ);
    pack_a_kernel<<<AFRAGS / 256, 256, 0, stream>>>(key,   ApK);
    pack_a_kernel<<<AFRAGS / 256, 256, 0, stream>>>(value, ApV);
    pack_b_kernel<<<4 * BFRAGS / 256, 256, 0, stream>>>(Wq, Wk, Wv, Wo, Bpack);

    proj_gemm_kernel<<<384, 256, 0, stream>>>(ApQ, ApK, ApV, Bpack,
                                              q_ws, k_ws, v_ws);
    scores_kernel<<<nslots / 256, 256, 0, stream>>>(pos_enc, pos_pad, rel_q, rel_k,
                                                    q_ws, k_ws, scores);
    softmax_kernel<<<B_ * H_, 256, 0, stream>>>(scores);
    scatter_kernel<<<(nslots / 256) * DK_, 256, 0, stream>>>(
        scores, pos_enc, pos_pad, v_ws, rel_v, heads);

    pack_heads_kernel<<<AFRAGS / 256, 256, 0, stream>>>(heads, ApH);
    out_gemm_kernel<<<128, 256, 0, stream>>>(
        ApH, Bpack + 3 * (size_t)BFRAGS * 16, out);
}
